// GNN_65807488909489
// MI455X (gfx1250) — matrix operand order repaired and checked by reading, not compiled, not run on hardware
//
#include <hip/hip_runtime.h>

#define NJ 512
#define NM 128
#define NF 8
#define NROWS (NJ * NM)   // 65536
#define HID 256
#define LDW 258           // LDS row stride in bf16 units (padded, gcd(129 dw,64)=1)
#define TROWS 64          // rows per workgroup

typedef __attribute__((ext_vector_type(16))) __bf16 bf16x16;
typedef __attribute__((ext_vector_type(8)))  float  floatx8;

union ABfrag { unsigned int u[8]; bf16x16 v; };
union Cfrag  { float f[8];        floatx8 v; };

__device__ __forceinline__ unsigned short f2bf(float f) {
    union { float f; unsigned int u; } a; a.f = f;
    unsigned int u = a.u;
    return (unsigned short)((u + 0x7FFFu + ((u >> 16) & 1u)) >> 16); // RNE
}

// ---------------- weight packing: f32 [din,dout] -> bf16 B-fragment tiles ----------------
// tile (ks, nt): 32 lanes x 16 bf16 contiguous; element (K=k%32, N=n%16):
//   lane = 16*(K/16) + N ; vreg = (K%16)/2 ; halfword = K%2
__global__ void k_pack(const float* __restrict__ W, unsigned short* __restrict__ dst,
                       int din, int dout, int Kp, int Np) {
    int idx = blockIdx.x * 256 + threadIdx.x;
    int tot = Kp * Np;
    if (idx >= tot) return;
    int k = idx / Np, n = idx - k * Np;
    float v = (k < din && n < dout) ? W[k * dout + n] : 0.0f;
    int ks = k >> 5, kt = k & 31, ntile = n >> 4, nn = n & 15;
    int lane = ((kt >> 4) << 4) | nn;
    int vreg = (kt & 15) >> 1, w = kt & 1;
    dst[(size_t)(ks * (Np >> 4) + ntile) * 512 + lane * 16 + vreg * 2 + w] = f2bf(v);
}

// ---------------- permutation inverse + pred/succ columns ----------------
__global__ void k_inv(const int* __restrict__ MM, int* __restrict__ inv) {
    int idx = blockIdx.x * 256 + threadIdx.x; // 65536
    int i = idx >> 7;
    inv[(i << 7) + MM[idx]] = idx & 127;
}
__global__ void k_predsucc(const int* __restrict__ MM, const int* __restrict__ inv,
                           int* __restrict__ pred, int* __restrict__ succ) {
    int idx = blockIdx.x * 256 + threadIdx.x;
    int i = idx >> 7;
    int s = MM[idx];
    pred[idx] = (s == 0)       ? -1 : inv[(i << 7) + s - 1];
    succ[idx] = (s == NM - 1)  ? -1 : inv[(i << 7) + s + 1];
}

// ---------------- column sums: sj[m*8+f] = sum_n x[n,m,f] ----------------
__global__ void k_colsum(const float* __restrict__ x, float* __restrict__ sj) {
    int idx = blockIdx.x * 256 + threadIdx.x; // 1024 = 128*8
    if (idx >= NM * NF) return;
    float s = 0.0f;
    for (int n = 0; n < NJ; ++n) s += x[n * (NM * NF) + idx];
    sj[idx] = s;
}
__global__ void k_total(const float* __restrict__ sj, float* __restrict__ a4) {
    int f = threadIdx.x;
    if (f < NF) {
        float s = 0.0f;
        for (int m = 0; m < NM; ++m) s += sj[m * NF + f];
        a4[f] = fmaxf(s, 0.0f);
    }
}

// ---------------- gathers -> bf16 rows padded to K=32 ----------------
__global__ void k_gather(const float* __restrict__ xcur, const int* __restrict__ pred,
                         const int* __restrict__ succ, const float* __restrict__ sj,
                         unsigned short* __restrict__ xp, unsigned short* __restrict__ xs,
                         unsigned short* __restrict__ s3) {
    int idx = blockIdx.x * 256 + threadIdx.x; // 65536
    int i = idx >> 7, j = idx & 127;
    int p = pred[idx], s = succ[idx];
    const float* base = xcur + (size_t)(i << 7) * NF;
    unsigned short* xpo = xp + (size_t)idx * 32;
    unsigned short* xso = xs + (size_t)idx * 32;
    unsigned short* s3o = s3 + (size_t)idx * 32;
    #pragma unroll
    for (int f = 0; f < NF; ++f) {
        float vp = (p < 0) ? 0.0f : base[p * NF + f];
        float vs = (s < 0) ? 0.0f : base[s * NF + f];
        float v3 = sj[j * NF + f] - base[(idx & 127) * NF + f];
        xpo[f] = f2bf(vp); xso[f] = f2bf(vs); s3o[f] = f2bf(v3);
    }
    #pragma unroll
    for (int f = NF; f < 32; ++f) { xpo[f] = 0; xso[f] = 0; s3o[f] = 0; }
}

// ---------------- feat concat [a1,a2,a3,a4,x,init] -> bf16 padded to 64 ----------------
__global__ void k_feat(const float* __restrict__ a1, const float* __restrict__ a2,
                       const float* __restrict__ a3, const float* __restrict__ a4,
                       const float* __restrict__ xcur, const float* __restrict__ xinit,
                       unsigned short* __restrict__ feat) {
    int idx = blockIdx.x * 256 + threadIdx.x; // 65536
    unsigned short* fo = feat + (size_t)idx * 64;
    size_t r8 = (size_t)idx * NF;
    #pragma unroll
    for (int f = 0; f < NF; ++f) {
        fo[f]      = f2bf(a1[r8 + f]);
        fo[8 + f]  = f2bf(a2[r8 + f]);
        fo[16 + f] = f2bf(a3[r8 + f]);
        fo[24 + f] = f2bf(a4[f]);
        fo[32 + f] = f2bf(xcur[r8 + f]);
        fo[40 + f] = f2bf(xinit[r8 + f]);
    }
    #pragma unroll
    for (int f = 48; f < 64; ++f) fo[f] = 0;
}

// ---------------- one fused GEMM layer over the 64-row LDS tile ----------------
template <int NT>
__device__ __forceinline__ void mlp_layer(unsigned short* lds,
                                          const unsigned short* __restrict__ Wp,
                                          const float* __restrict__ bias,
                                          int ksteps, int nct_total, int nvalid,
                                          bool dorelu, float* __restrict__ gout, int row0) {
    const int tid  = threadIdx.x;
    const int wave = tid >> 5;
    const int lane = tid & 31;
    const int mrow = lane & 15;  // A row / C column within tile
    const int half = lane >> 4;

    int rt, ct0, ctstep, nt;
    if (NT > 1) { rt = wave & 3; ct0 = wave >> 2; ctstep = 2; nt = NT; }
    else        { rt = wave;     ct0 = 0;         ctstep = 1; nt = (wave < 4) ? 1 : 0; }

    Cfrag acc[NT];
    #pragma unroll
    for (int t = 0; t < NT; ++t)
        #pragma unroll
        for (int v = 0; v < 8; ++v) acc[t].f[v] = 0.0f;

    for (int ks = 0; ks < ksteps; ++ks) {
        ABfrag a;
        if (nt > 0) {
            // A 16x32 bf16 fragment (contiguous K): lanes0-15 K 0..15, lanes16-31 K 16..31
            const unsigned short* ap = lds + (rt * 16 + mrow) * LDW + ks * 32 + 16 * half;
            #pragma unroll
            for (int v = 0; v < 4; ++v) a.u[v] = *(const unsigned int*)(ap + 2 * v);
            #pragma unroll
            for (int v = 0; v < 4; ++v) a.u[4 + v] = *(const unsigned int*)(ap + 8 + 2 * v);
        }
        #pragma unroll
        for (int t = 0; t < NT; ++t) {
            if (t < nt) { // wave-uniform predicate: EXEC stays all-1s for WMMA
                int ct = ct0 + t * ctstep;
                const uint4* bp = (const uint4*)Wp + (size_t)(ks * nct_total + ct) * 32 + lane * 2;
                uint4 b0 = bp[0], b1 = bp[1];
                ABfrag b;
                b.u[0] = b0.x; b.u[1] = b0.y; b.u[2] = b0.z; b.u[3] = b0.w;
                b.u[4] = b1.x; b.u[5] = b1.y; b.u[6] = b1.z; b.u[7] = b1.w;
                acc[t].v = __builtin_amdgcn_wmma_f32_16x16x32_bf16(
                    false, a.v, false, b.v, (short)0, acc[t].v, false, false);
            }
        }
    }
    __syncthreads(); // all LDS reads done before overwrite
    if (gout == nullptr) {
        #pragma unroll
        for (int t = 0; t < NT; ++t) {
            if (t < nt) {
                int N = (ct0 + t * ctstep) * 16 + mrow;
                float bv = bias[N];
                #pragma unroll
                for (int v = 0; v < 8; ++v) {
                    float xv = acc[t].f[v] + bv;
                    if (dorelu) xv = fmaxf(xv, 0.0f);
                    lds[(rt * 16 + v + 8 * half) * LDW + N] = f2bf(xv);
                }
            }
        }
    } else {
        if (nt > 0 && mrow < nvalid) {
            float bv = bias[mrow];
            #pragma unroll
            for (int v = 0; v < 8; ++v) {
                float xv = acc[0].f[v] + bv;
                if (dorelu) xv = fmaxf(xv, 0.0f);
                gout[(size_t)(row0 + rt * 16 + v + 8 * half) * NF + mrow] = xv;
            }
        }
    }
    __syncthreads(); // stores visible before next layer reads
}

// ---------------- fused 4-layer MLP over 64 rows per block ----------------
__global__ __launch_bounds__(256) void k_mlp(
    const unsigned short* __restrict__ in, int din_pad,
    const unsigned short* __restrict__ Wp1, const float* __restrict__ b1,
    const unsigned short* __restrict__ Wp2, const float* __restrict__ b2,
    const unsigned short* __restrict__ Wp3, const float* __restrict__ b3,
    const unsigned short* __restrict__ Wp4, const float* __restrict__ b4,
    float* __restrict__ out, int final_relu) {
    __shared__ unsigned short lds[TROWS * LDW]; // 33 KB
    const int row0 = blockIdx.x * TROWS;
    const int tid = threadIdx.x;
    const int cpr = din_pad >> 1; // dwords per row
    const int ndw = TROWS * cpr;
    for (int i = tid; i < ndw; i += 256) {
        int r = i / cpr, c = i - r * cpr;
        *(unsigned int*)(lds + r * LDW + c * 2) =
            *(const unsigned int*)(in + (size_t)(row0 + r) * din_pad + c * 2);
    }
    __syncthreads();
    mlp_layer<8>(lds, Wp1, b1, din_pad >> 5, 16, 256, true, nullptr, row0);
    mlp_layer<8>(lds, Wp2, b2, 8, 16, 256, true, nullptr, row0);
    mlp_layer<8>(lds, Wp3, b3, 8, 16, 256, true, nullptr, row0);
    mlp_layer<1>(lds, Wp4, b4, 8, 1, NF, final_relu != 0, out, row0);
}

extern "C" void kernel_launch(void* const* d_in, const int* in_sizes, int n_in,
                              void* d_out, int out_size, void* d_ws, size_t ws_size,
                              hipStream_t stream) {
    const float* x0 = (const float*)d_in[0];
    const int* MM = (const int*)d_in[1];
    const float* W[4][4]; const float* B[4][4];
    for (int m = 0; m < 4; ++m)
        for (int l = 0; l < 4; ++l) {
            W[m][l] = (const float*)d_in[3 + m * 8 + l * 2];
            B[m][l] = (const float*)d_in[3 + m * 8 + l * 2 + 1];
        }

    char* base = (char*)d_ws; size_t off = 0;
    auto alloc = [&](size_t bytes) -> void* {
        void* p = base + off; off += (bytes + 255) & ~(size_t)255; return p;
    };
    int*   inv  = (int*)alloc((size_t)NROWS * 4);
    int*   pred = (int*)alloc((size_t)NROWS * 4);
    int*   succ = (int*)alloc((size_t)NROWS * 4);
    float* sj   = (float*)alloc((size_t)NM * NF * 4);
    float* a4b  = (float*)alloc((size_t)NF * 4);
    unsigned short* xp = (unsigned short*)alloc((size_t)NROWS * 32 * 2);
    unsigned short* xs = (unsigned short*)alloc((size_t)NROWS * 32 * 2);
    unsigned short* s3 = (unsigned short*)alloc((size_t)NROWS * 32 * 2);
    float* a1 = (float*)alloc((size_t)NROWS * NF * 4);
    float* a2 = (float*)alloc((size_t)NROWS * NF * 4);
    float* a3 = (float*)alloc((size_t)NROWS * NF * 4);
    unsigned short* feat = (unsigned short*)alloc((size_t)NROWS * 64 * 2);
    float* xb0 = (float*)alloc((size_t)NROWS * NF * 4);
    float* xb1 = (float*)alloc((size_t)NROWS * NF * 4);

    unsigned short* Wp[4][4];
    for (int m = 0; m < 4; ++m)
        for (int l = 0; l < 4; ++l) {
            int Kp = (l == 0) ? ((m == 3) ? 64 : 32) : 256;
            int Np = (l == 3) ? 16 : 256;
            Wp[m][l] = (unsigned short*)alloc((size_t)Kp * Np * 2);
        }

    // weight packing (every launch: deterministic, cheap ~1 MB)
    for (int m = 0; m < 4; ++m)
        for (int l = 0; l < 4; ++l) {
            int din  = (l == 0) ? ((m == 3) ? 48 : 8) : 256;
            int dout = (l == 3) ? 8 : 256;
            int Kp   = (l == 0) ? ((m == 3) ? 64 : 32) : 256;
            int Np   = (l == 3) ? 16 : 256;
            int tot  = Kp * Np;
            hipLaunchKernelGGL(k_pack, dim3((tot + 255) / 256), dim3(256), 0, stream,
                               W[m][l], Wp[m][l], din, dout, Kp, Np);
        }

    hipLaunchKernelGGL(k_inv, dim3(NROWS / 256), dim3(256), 0, stream, MM, inv);
    hipLaunchKernelGGL(k_predsucc, dim3(NROWS / 256), dim3(256), 0, stream, MM, inv, pred, succ);

    const float* xcur = x0;
    for (int sweep = 0; sweep < 3; ++sweep) {
        float* xnext = (sweep == 2) ? (float*)d_out : ((sweep == 0) ? xb0 : xb1);
        hipLaunchKernelGGL(k_colsum, dim3(4), dim3(256), 0, stream, xcur, sj);
        hipLaunchKernelGGL(k_total, dim3(1), dim3(32), 0, stream, sj, a4b);
        hipLaunchKernelGGL(k_gather, dim3(NROWS / 256), dim3(256), 0, stream,
                           xcur, pred, succ, sj, xp, xs, s3);
        hipLaunchKernelGGL(k_mlp, dim3(NROWS / TROWS), dim3(256), 0, stream,
                           xp, 32, Wp[0][0], B[0][0], Wp[0][1], B[0][1],
                           Wp[0][2], B[0][2], Wp[0][3], B[0][3], a1, 1);
        hipLaunchKernelGGL(k_mlp, dim3(NROWS / TROWS), dim3(256), 0, stream,
                           xs, 32, Wp[1][0], B[1][0], Wp[1][1], B[1][1],
                           Wp[1][2], B[1][2], Wp[1][3], B[1][3], a2, 1);
        hipLaunchKernelGGL(k_mlp, dim3(NROWS / TROWS), dim3(256), 0, stream,
                           s3, 32, Wp[2][0], B[2][0], Wp[2][1], B[2][1],
                           Wp[2][2], B[2][2], Wp[2][3], B[2][3], a3, 1);
        hipLaunchKernelGGL(k_feat, dim3(NROWS / 256), dim3(256), 0, stream,
                           a1, a2, a3, a4b, xcur, x0, feat);
        hipLaunchKernelGGL(k_mlp, dim3(NROWS / TROWS), dim3(256), 0, stream,
                           feat, 64, Wp[3][0], B[3][0], Wp[3][1], B[3][1],
                           Wp[3][2], B[3][2], Wp[3][3], B[3][3], xnext, 0);
        xcur = xnext;
    }
}
